// GrowthRotation_3616362463457
// MI455X (gfx1250) — matrix operand order repaired and checked by reading, not compiled, not run on hardware
//
#include <hip/hip_runtime.h>
#include <hip/hip_bf16.h>
#include <math.h>

// ---------------- problem constants (match reference) ----------------
#define RMINC 10
#define RMAXC 50
#define GAPC  40
#define RC    41            // R
#define PC    128           // P
#define NSTEP 32            // N
#define BATCH 16384         // B
#define MCC   10000         // MC
#define DTC   (1.0f/32.0f)
#define SQRT_DT 0.17677669529663687f
#define SQ2D  1.4142135623730951f     // sqrt(2*DD)
#define PI_F  3.14159265358979323846f

// per-step packed-weight blob: [W2 (64x64) | W3 (64x128)] in bf16 fragments
#define WBLOB_U16 12288     // 4096 + 8192
#define W3_OFF    4096

typedef unsigned int   u32;
typedef unsigned short u16;

typedef __attribute__((ext_vector_type(8)))  float  v8f;
typedef __attribute__((ext_vector_type(16))) __bf16 v16bf;

// gcc-style int4 vector in explicit address spaces for the async-LDS builtin
typedef int gr_v4i __attribute__((vector_size(16)));
typedef __attribute__((address_space(1))) gr_v4i gr_as1_v4i;   // global
typedef __attribute__((address_space(3))) gr_v4i gr_as3_v4i;   // LDS

union Frag32B { uint4 q[2]; u32 u[8]; v16bf v; };

#if __has_builtin(__builtin_amdgcn_global_load_async_to_lds_b128)
#define GR_ASYNC_LDS 1
#else
#define GR_ASYNC_LDS 0
#endif

// ---------------- helpers ----------------
__device__ __forceinline__ float gr_rcp(float x) {
#if __has_builtin(__builtin_amdgcn_rcpf)
  return __builtin_amdgcn_rcpf(x);
#else
  return 1.0f / x;
#endif
}
__device__ __forceinline__ float gr_rsq(float x) {
#if __has_builtin(__builtin_amdgcn_rsqf)
  return __builtin_amdgcn_rsqf(x);
#else
  return rsqrtf(x);
#endif
}

__device__ __forceinline__ float ftanh(float x) {
  x = fminf(fmaxf(x, -15.f), 15.f);
  float e = __expf(2.f * x);                 // v_exp_f32
  return (e - 1.f) * gr_rcp(e + 1.f);        // v_rcp_f32, no div macro
}

__device__ __forceinline__ u16 f2bf(float f) {
  u32 a = __float_as_uint(f);
  a = (a + 0x7FFFu + ((a >> 16) & 1u)) >> 16;   // round-to-nearest-even
  return (u16)a;
}
__device__ __forceinline__ u32 f2bf_pk(float lo, float hi) {
  return (u32)f2bf(lo) | ((u32)f2bf(hi) << 16);
}

__device__ __forceinline__ v8f wmma_bf16(v16bf a, v16bf b, v8f c) {
  // D(16x16,f32) = A(16x32,bf16) * B(32x16,bf16) + C
  return __builtin_amdgcn_wmma_f32_16x16x32_bf16(false, a, false, b, (short)0, c,
                                                 false, false);
}

// B-fragment: packed so each lane's 16 bf16 values are 32 contiguous bytes.
// Works for both global (global_load_b128) and LDS (ds_load_b128) provenance.
__device__ __forceinline__ v16bf load_bfrag(const u16* p, int frag, int lane) {
  const uint4* q = (const uint4*)(p + (size_t)frag * 512 + lane * 16);
  Frag32B f; f.q[0] = q[0]; f.q[1] = q[1];
  return f.v;
}

// sum across the 16 lanes of each half-wave (masks < 16 keep halves separate)
__device__ __forceinline__ float redux16(float v) {
  v += __shfl_xor(v, 1, 32);
  v += __shfl_xor(v, 2, 32);
  v += __shfl_xor(v, 4, 32);
  v += __shfl_xor(v, 8, 32);
  return v;
}

// Stage one 24KB weight blob (global, fragment-packed) into LDS.
// Uses CDNA5 async global->LDS DMA when the builtin is available.
__device__ __forceinline__ void gr_stage24k(const u16* __restrict__ gsrc,
                                            u16* __restrict__ ldst, int tid) {
  const uint4* s = (const uint4*)gsrc;
  uint4* d = (uint4*)ldst;
#if GR_ASYNC_LDS
#pragma unroll
  for (int i = 0; i < 6; ++i) {
    int idx = tid + i * 256;      // 1536 x 16B total
    __builtin_amdgcn_global_load_async_to_lds_b128(
        (gr_as1_v4i*)(s + idx),
        (gr_as3_v4i*)(d + idx), 0, 0);
  }
#if __has_builtin(__builtin_amdgcn_s_wait_asynccnt)
  __builtin_amdgcn_s_wait_asynccnt(0);
#else
  asm volatile("s_wait_asynccnt 0" ::: "memory");
#endif
#else
#pragma unroll
  for (int i = 0; i < 6; ++i) {
    int idx = tid + i * 256;
    d[idx] = s[idx];
  }
#endif
  __syncthreads();
}

// ---------------- 3 -> 64 -> 64 -> 128 tanh MLP for a 16-row tile ----------------
// Layer1 on VALU written directly into WMMA A-fragment layout; layers 2/3 WMMA bf16.
__device__ __forceinline__ void run_mlp(
    const float* __restrict__ W1, const float* __restrict__ b1,
    const u16*   __restrict__ W2p, const float* __restrict__ b2,
    const u16*   __restrict__ W3p, const float* __restrict__ b3,
    float x0, float x1, float x2,
    u16* __restrict__ shrow,       // per-wave 16x64 bf16 staging (row-major)
    int h, int ln, int lane,
    v8f out[8])
{
  // ---- layer 1 (K=3): lane L holds row ln; K per ISA 16-bit A layout ----
  Frag32B A0, A1;
#pragma unroll
  for (int j = 0; j < 8; ++j) {
    int kb = (j < 4) ? (8 * h + 2 * j) : (16 + 8 * h + 2 * (j - 4));
    float v00, v01, v10, v11;
    {
      int k = kb;
      v00 = ftanh(fmaf(x0, W1[k], fmaf(x1, W1[64 + k], fmaf(x2, W1[128 + k], b1[k]))));
      k = kb + 1;
      v01 = ftanh(fmaf(x0, W1[k], fmaf(x1, W1[64 + k], fmaf(x2, W1[128 + k], b1[k]))));
      k = kb + 32;
      v10 = ftanh(fmaf(x0, W1[k], fmaf(x1, W1[64 + k], fmaf(x2, W1[128 + k], b1[k]))));
      k = kb + 33;
      v11 = ftanh(fmaf(x0, W1[k], fmaf(x1, W1[64 + k], fmaf(x2, W1[128 + k], b1[k]))));
    }
    A0.u[j] = f2bf_pk(v00, v01);
    A1.u[j] = f2bf_pk(v10, v11);
  }

  // ---- layer 2: h2(16x64) = h1(16x64) @ W2(64x64) + b2, tanh ----
#pragma unroll
  for (int nt = 0; nt < 4; ++nt) {
    float bv = b2[nt * 16 + ln];
    v8f c;
#pragma unroll
    for (int i = 0; i < 8; ++i) c[i] = bv;
    c = wmma_bf16(A0.v, load_bfrag(W2p, 0 * 4 + nt, lane), c);
    c = wmma_bf16(A1.v, load_bfrag(W2p, 1 * 4 + nt, lane), c);
    // C/D layout: lane holds rows 8h..8h+7 of column 16*nt+ln -> stage to LDS
#pragma unroll
    for (int v = 0; v < 8; ++v)
      shrow[(8 * h + v) * 64 + nt * 16 + ln] = f2bf(ftanh(c[v]));
  }
  __syncthreads();

  // ---- rebuild A fragments from LDS (transpose C-layout -> A-layout) ----
  Frag32B B0, B1;
  const u16* rp = shrow + ln * 64;
  B0.q[0] = *(const uint4*)(rp + 8 * h);
  B0.q[1] = *(const uint4*)(rp + 16 + 8 * h);
  B1.q[0] = *(const uint4*)(rp + 32 + 8 * h);
  B1.q[1] = *(const uint4*)(rp + 48 + 8 * h);

  // ---- layer 3: u(16x128) = h2(16x64) @ W3(64x128) + b3 ----
#pragma unroll
  for (int nt = 0; nt < 8; ++nt) {
    float bv = b3[nt * 16 + ln];
    v8f c;
#pragma unroll
    for (int i = 0; i < 8; ++i) c[i] = bv;
    c = wmma_bf16(B0.v, load_bfrag(W3p, 0 * 8 + nt, lane), c);
    c = wmma_bf16(B1.v, load_bfrag(W3p, 1 * 8 + nt, lane), c);
    out[nt] = c;
  }
  __syncthreads();   // shrow (and any LDS weight buffer) reusable afterwards
}

// ---------------- precompute kernels ----------------

// jump tables: normalized cumulative measure (R x 80), cr (R), cfr (R)
__global__ void gr_tables_kernel(float* __restrict__ jmn, float* __restrict__ cr,
                                 float* __restrict__ cfr) {
  int r = threadIdx.x;
  if (r >= RC) return;
  float rv = (float)(RMINC + r);
  float run = 0.f, ls = 0.f, ms = 0.f;
  for (int k = 1; k <= GAPC; ++k) {
    float v = (rv + (float)k <= (float)RMAXC) ? 1.0f / (float)(k * k) : 0.f;
    ls += v; run += v; jmn[r * 80 + k - 1] = run;
  }
  for (int k = 1; k <= GAPC; ++k) {
    float v = (rv - (float)k >= (float)RMINC) ? 1.0f / (float)(k * k) : 0.f;
    ms += v; run += v; jmn[r * 80 + GAPC + k - 1] = run;
  }
  cr[r]  = run;
  cfr[r] = ls + ms - run;
  float inv = 1.f / run;
  for (int j = 0; j < 80; ++j) jmn[r * 80 + j] *= inv;
}

// mc_jump (R x P) via per-r histogram of jump sizes over MC samples
__global__ void gr_mcjump_kernel(const float* __restrict__ u_mc,
                                 const float* __restrict__ jump_l,
                                 const float* __restrict__ jmn,
                                 const float* __restrict__ cr,
                                 float* __restrict__ mcj) {
  __shared__ int   hist[80];
  __shared__ float jmr[80];
  int r = blockIdx.x, t = threadIdx.x;   // 128 threads
  if (t < 80) { hist[t] = 0; jmr[t] = jmn[r * 80 + t]; }
  __syncthreads();
  for (int q = t; q < MCC; q += 128) {
    float u = u_mc[q * RC + r];
    int cnt = 0;
    for (int j = 0; j < 80; ++j) cnt += (u < jmr[j]) ? 1 : 0;
    int ind = 80 - cnt;           // bin: 0..39 => +k, 40..79 => -k
    atomicAdd(&hist[ind], 1);
  }
  __syncthreads();
  float acc = 0.f;
  for (int kb = 0; kb < 80; ++kb)
    acc += (float)hist[kb] * jump_l[kb * PC + t];   // (2,40,128) flat == bin*128
  mcj[r * PC + t] = acc * cr[r] * (1.0f / (float)MCC);
}

// pack f32 (K x Nc) weights (count matrices) into bf16 WMMA B-fragment layout
// dst element index = cm * dstStride + dstOff + e
__global__ void gr_pack_bf16_kernel(const float* __restrict__ src, u16* __restrict__ dst,
                                    int K, int Nc, int count, long dstOff, long dstStride) {
  int NT = Nc >> 4;
  long total = (long)K * Nc;
  long tot = total * count;
  for (long idx = blockIdx.x * (long)blockDim.x + threadIdx.x; idx < tot;
       idx += (long)gridDim.x * blockDim.x) {
    long cm = idx / total;
    int e = (int)(idx - cm * total);
    int frag = e >> 9, t = e & 511;
    int L = t >> 4, q = t & 15;
    int j = q >> 1, s = q & 1;
    int kt = frag / NT, nt = frag - kt * NT;
    int hh = L >> 4;
    int col = nt * 16 + (L & 15);
    int k = kt * 32 + ((j < 4) ? (8 * hh + 2 * j + s) : (16 + 8 * hh + 2 * (j - 4) + s));      // ISA 16-bit B (32x16) layout
    u32 a = __float_as_uint(src[cm * total + (long)k * Nc + col]);
    a = (a + 0x7FFFu + ((a >> 16) & 1u)) >> 16;
    dst[cm * dstStride + dstOff + e] = (u16)a;
  }
}

// ---------------- main simulation kernel ----------------
__global__ __launch_bounds__(256, 1) void gr_sim_kernel(
    const int*   __restrict__ rt0,
    const float* __restrict__ xt0,
    const float* __restrict__ dBt,
    const float* __restrict__ u_jump,
    const float* __restrict__ u_size,
    const float* __restrict__ uW1, const float* __restrict__ ub1,
    const float* __restrict__ ub2, const float* __restrict__ ub3,
    const float* __restrict__ uE,
    const float* __restrict__ guW1, const float* __restrict__ gub1,
    const float* __restrict__ gub2, const float* __restrict__ gub3,
    const float* __restrict__ guE,
    const float* __restrict__ jxW1, const float* __restrict__ jxb1,
    const float* __restrict__ jxb2, const float* __restrict__ jxb3,
    const float* __restrict__ jump_r, const float* __restrict__ jump_l,
    const float* __restrict__ jmn, const float* __restrict__ crt,
    const float* __restrict__ cfrt, const float* __restrict__ mcj,
    const u16* __restrict__ uW2p, const u16* __restrict__ uW3p,
    const u16* __restrict__ gblob, const u16* __restrict__ jblob,
    float* __restrict__ outp)
{
  __shared__ __align__(16) u16 sh_w[WBLOB_U16];   // 24KB staged step weights [W2|W3]
  __shared__ __align__(16) u16 sh_h[8][1024];     // per-wave 16x64 bf16 staging
  __shared__ float sh_x[8][16][3];
  __shared__ int   sh_ridx[8][16];
  __shared__ int   sh_drt[8][16];
  __shared__ float sh_res[8][16][4];

  const int tid = threadIdx.x;
  const int w = tid >> 5, lane = tid & 31;
  const int h = lane >> 4, ln = lane & 15;
  const int gb = blockIdx.x * 128 + w * 16 + ln;   // batch row of this lane's tile row
  const bool owner = (lane < 16);

  float xa = 0.f, xb = 0.f, xc = 0.f;              // xt
  float xi0 = 0.f, xi1 = 0.f, xi2 = 0.f;           // xt_in
  float g00 = 1, g01 = 0, g02 = 0, g10 = 0, g11 = 1, g12 = 0, g20 = 0, g21 = 0, g22 = 1;
  float u_pre = 0.f, fun = 0.f;
  int rt = RMINC;

  if (owner) {
    rt = rt0[gb];
    xa = xt0[gb * 3 + 0]; xb = xt0[gb * 3 + 1]; xc = xt0[gb * 3 + 2];
    xi0 = xa; xi1 = xb; xi2 = xc;
    sh_x[w][ln][0] = xi0; sh_x[w][ln][1] = xi1; sh_x[w][ln][2] = xi2;
    sh_ridx[w][ln] = rt - RMINC;
  }
  __syncthreads();

  // ===== initial u_pre = tnn(ridx0, xt0; u-network, uE) -- weights from global =====
  {
    float x0 = sh_x[w][ln][0], x1 = sh_x[w][ln][1], x2 = sh_x[w][ln][2];
    int rix[8];
#pragma unroll
    for (int v = 0; v < 8; ++v) rix[v] = sh_ridx[w][8 * h + v];
    v8f out[8];
    run_mlp(uW1, ub1, uW2p, ub2, uW3p, ub3, x0, x1, x2, sh_h[w], h, ln, lane, out);
    float p[8];
#pragma unroll
    for (int v = 0; v < 8; ++v) p[v] = 0.f;
#pragma unroll
    for (int nt = 0; nt < 8; ++nt) {
      int c = nt * 16 + ln;
#pragma unroll
      for (int v = 0; v < 8; ++v)
        p[v] = fmaf(out[nt][v], uE[rix[v] * PC + c], p[v]);
    }
#pragma unroll
    for (int v = 0; v < 8; ++v) {
      float s = redux16(p[v]);
      if (ln == 0) sh_res[w][8 * h + v][0] = s;
    }
    __syncthreads();
    if (owner) u_pre = sh_res[w][ln][0];
    __syncthreads();
  }

  // ===== time stepping =====
#pragma unroll 1
  for (int step = 0; step < NSTEP; ++step) {
    const float* gW1 = guW1 + step * 192;
    const float* gb1 = gub1 + step * 64;
    const float* gb2 = gub2 + step * 64;
    const float* gb3 = gub3 + step * 128;
    const float* gE  = guE + (size_t)step * RC * PC * 3;
    const float* jW1 = jxW1 + step * 192;
    const float* jb1 = jxb1 + step * 64;
    const float* jb2 = jxb2 + step * 64;
    const float* jb3 = jxb3 + step * 128;
    const u16* gWstep = gblob + (size_t)step * WBLOB_U16;
    const u16* jWstep = jblob + (size_t)step * WBLOB_U16;

    int ridx = 0, drt = 0;
    if (owner) {
      ridx = rt - RMINC;
      sh_ridx[w][ln] = ridx;
      sh_x[w][ln][0] = xi0; sh_x[w][ln][1] = xi1; sh_x[w][ln][2] = xi2;
      float uj = u_jump[step * BATCH + gb];
      if (uj < crt[ridx] * DTC) {
        float us = u_size[step * BATCH + gb];
        const float* jm = jmn + ridx * 80;
        int cnt = 0;
        for (int j = 0; j < 80; ++j) cnt += (us < jm[j]) ? 1 : 0;
        int ind = 80 - cnt;
        drt = (ind < GAPC) ? (ind + 1) : -(ind - GAPC + 1);
      }
      sh_drt[w][ln] = drt;
    }
    __syncthreads();

    float x0 = sh_x[w][ln][0], x1 = sh_x[w][ln][1], x2 = sh_x[w][ln][2];
    int rix[8], drv[8];
#pragma unroll
    for (int v = 0; v < 8; ++v) {
      rix[v] = sh_ridx[w][8 * h + v];
      drv[v] = sh_drt[w][8 * h + v];
    }

    // ---- g-network: stage weights to LDS, then gu = tnn(ridx, xt_in) ----
    gr_stage24k(gWstep, sh_w, tid);
    {
      v8f out[8];
      run_mlp(gW1, gb1, sh_w, gb2, sh_w + W3_OFF, gb3, x0, x1, x2,
              sh_h[w], h, ln, lane, out);
      float p0[8], p1[8], p2[8];
#pragma unroll
      for (int v = 0; v < 8; ++v) { p0[v] = 0.f; p1[v] = 0.f; p2[v] = 0.f; }
#pragma unroll
      for (int nt = 0; nt < 8; ++nt) {
        int c = nt * 16 + ln;
#pragma unroll
        for (int v = 0; v < 8; ++v) {
          const float* e = gE + ((size_t)rix[v] * PC + c) * 3;
          float uv = out[nt][v];
          p0[v] = fmaf(uv, e[0], p0[v]);
          p1[v] = fmaf(uv, e[1], p1[v]);
          p2[v] = fmaf(uv, e[2], p2[v]);
        }
      }
#pragma unroll
      for (int v = 0; v < 8; ++v) {
        float s0 = redux16(p0[v]);
        float s1 = redux16(p1[v]);
        float s2 = redux16(p2[v]);
        if (ln == 0) {
          sh_res[w][8 * h + v][0] = s0;
          sh_res[w][8 * h + v][1] = s1;
          sh_res[w][8 * h + v][2] = s2;
        }
      }
    }

    // ---- jx-network: jmp = sum_p jx * jump_r[ridx] * (jl - mc_jump[ridx]*DT) ----
    gr_stage24k(jWstep, sh_w, tid);
    // prefetch next step's weight blobs while we compute
    if (step + 1 < NSTEP) {
      __builtin_prefetch((const char*)(gWstep + WBLOB_U16) + tid * 96, 0, 1);
      __builtin_prefetch((const char*)(jWstep + WBLOB_U16) + tid * 96, 0, 1);
    }
    {
      v8f out[8];
      run_mlp(jW1, jb1, sh_w, jb2, sh_w + W3_OFF, jb3, x0, x1, x2,
              sh_h[w], h, ln, lane, out);
      float p[8];
#pragma unroll
      for (int v = 0; v < 8; ++v) p[v] = 0.f;
#pragma unroll
      for (int nt = 0; nt < 8; ++nt) {
        int c = nt * 16 + ln;
#pragma unroll
        for (int v = 0; v < 8; ++v) {
          float jr = jump_r[rix[v] * PC + c];
          float mj = mcj[rix[v] * PC + c];
          float jl = 0.f;
          int d = drv[v];
          if (d != 0) {
            int ad = (d < 0) ? (-d - 1) : (d - 1);
            if (ad > GAPC - 1) ad = GAPC - 1;
            int sg = (d < 0) ? 1 : 0;
            jl = jump_l[(sg * GAPC + ad) * PC + c];
          }
          p[v] = fmaf(out[nt][v], jr * (jl - mj * DTC), p[v]);
        }
      }
#pragma unroll
      for (int v = 0; v < 8; ++v) {
        float s = redux16(p[v]);
        if (ln == 0) sh_res[w][8 * h + v][3] = s;
      }
    }
    __syncthreads();

    // ---- per-row geometry update (owner lanes) ----
    if (owner) {
      float gu0 = sh_res[w][ln][0], gu1 = sh_res[w][ln][1], gu2 = sh_res[w][ln][2];
      float jmp = sh_res[w][ln][3];
      float rnv = sqrtf(xa * xa + xb * xb + xc * xc);
      float a = fminf(fmaxf(xc / rnv, -1.f), 1.f);
      // theta = acos(a) - pi/2  =>  cos(theta) = sqrt(1-a^2), sin(theta) = -a
      float ct = sqrtf(fmaxf(1.f - a * a, 0.f));
      float st = -a;
      // phi = atan2(xb, xa)     =>  cos(phi) = xa/h, sin(phi) = xb/h
      float xy2 = xa * xa + xb * xb;
      float cp, sp;
      if (xy2 > 1e-30f) {
        float invh = gr_rsq(xy2);
        cp = xa * invh; sp = xb * invh;
      } else {
        cp = 1.f; sp = 0.f;
      }
      float T00 = cp * ct, T01 = -sp, T02 = cp * st;
      float T10 = sp * ct, T11 = cp, T12 = sp * st;
      float T20 = -st, T21 = 0.f, T22 = ct;
      // v = gu . gt . Tinv  (only components 1,2 needed)
      float w0 = gu0 * g00 + gu1 * g10 + gu2 * g20;
      float w1 = gu0 * g01 + gu1 * g11 + gu2 * g21;
      float w2 = gu0 * g02 + gu1 * g12 + gu2 * g22;
      float v1 = w0 * T01 + w1 * T11 + w2 * T21;
      float v2 = w0 * T02 + w1 * T12 + w2 * T22;
      float rt_f = (float)rt;
      float dB0 = dBt[((size_t)step * BATCH + gb) * 2 + 0] * SQRT_DT;
      float dB1 = dBt[((size_t)step * BATCH + gb) * 2 + 1] * SQRT_DT;
      float s2r = SQ2D / rt_f;
      u_pre += __expf(-fun) * (s2r * (-v2 * dB0 + v1 * dB1) + jmp);
      fun += cfrt[ridx] * DTC;
      float dX20 = s2r * dB0, dX21 = s2r * dB1;
      float thd = dX20 + 0.5f * PI_F, phd = dX21;
      float sthd = sinf(thd), cthd = cosf(thd);
      float cphd = cosf(phd), sphd = sinf(phd);
      float d30 = sthd * cphd - 1.f, d31 = sthd * sphd, d32 = cthd;
      float dX0 = T00 * d30 + T01 * d31 + T02 * d32;
      float dX1 = T10 * d30 + T11 * d31 + T12 * d32;
      float dX2 = T20 * d30 + T21 * d31 + T22 * d32;
      xa += dX0; xb += dX1; xc += dX2;
      float di0 = g00 * dX0 + g01 * dX1 + g02 * dX2;
      float di1 = g10 * dX0 + g11 * dX1 + g12 * dX2;
      float di2 = g20 * dX0 + g21 * dX1 + g22 * dX2;
      xi0 += di0; xi1 += di1; xi2 += di2;
      rt += drt;
      if (xi2 < 0.f) {            // reflect: negate z of xt_in, negate row 2 of gt
        xi2 = -xi2;
        g20 = -g20; g21 = -g21; g22 = -g22;
      }
    }
    __syncthreads();
  }

  if (owner) {
    float rr = xi0 * xi0 + xi1 * xi1 + xi2 * xi2;
    outp[gb] = u_pre;
    outp[BATCH + gb] = rr / (float)rt * __expf(-fun);
  }
}

// ---------------- host-side launch ----------------
extern "C" void kernel_launch(void* const* d_in, const int* in_sizes, int n_in,
                              void* d_out, int out_size, void* d_ws, size_t ws_size,
                              hipStream_t stream) {
  (void)in_sizes; (void)n_in; (void)out_size; (void)ws_size;
  const int*   rt0    = (const int*)  d_in[1];
  const float* xt0    = (const float*)d_in[2];
  const float* dBt    = (const float*)d_in[3];
  const float* u_jump = (const float*)d_in[4];
  const float* u_size = (const float*)d_in[5];
  const float* u_mc   = (const float*)d_in[6];
  const float* uW1    = (const float*)d_in[7];
  const float* ub1    = (const float*)d_in[8];
  const float* uW2    = (const float*)d_in[9];
  const float* ub2    = (const float*)d_in[10];
  const float* uW3    = (const float*)d_in[11];
  const float* ub3    = (const float*)d_in[12];
  const float* uE     = (const float*)d_in[13];
  const float* guW1   = (const float*)d_in[14];
  const float* gub1   = (const float*)d_in[15];
  const float* guW2   = (const float*)d_in[16];
  const float* gub2   = (const float*)d_in[17];
  const float* guW3   = (const float*)d_in[18];
  const float* gub3   = (const float*)d_in[19];
  const float* guE    = (const float*)d_in[20];
  const float* jxW1   = (const float*)d_in[21];
  const float* jxb1   = (const float*)d_in[22];
  const float* jxW2   = (const float*)d_in[23];
  const float* jxb2   = (const float*)d_in[24];
  const float* jxW3   = (const float*)d_in[25];
  const float* jxb3   = (const float*)d_in[26];
  const float* jump_r = (const float*)d_in[27];
  const float* jump_l = (const float*)d_in[28];

  // workspace layout
  float* ws_f = (float*)d_ws;
  float* jmn = ws_f;                 // R*80
  float* cr  = jmn + RC * 80;        // R
  float* cfr = cr + RC;              // R
  float* mcj = cfr + RC;             // R*128
  u16* pw    = (u16*)((char*)d_ws + 34816);   // 16B aligned, past f32 tables
  u16* uW2p  = pw;                            // 4096
  u16* uW3p  = pw + 4096;                     // 8192
  u16* gblob = pw + 12288;                    // NSTEP * WBLOB_U16
  u16* jblob = gblob + (size_t)NSTEP * WBLOB_U16;

  gr_tables_kernel<<<1, 64, 0, stream>>>(jmn, cr, cfr);
  gr_pack_bf16_kernel<<<16,  256, 0, stream>>>(uW2,  uW2p,  64, 64,  1,     0,      4096);
  gr_pack_bf16_kernel<<<32,  256, 0, stream>>>(uW3,  uW3p,  64, 128, 1,     0,      8192);
  gr_pack_bf16_kernel<<<256, 256, 0, stream>>>(guW2, gblob, 64, 64,  NSTEP, 0,      WBLOB_U16);
  gr_pack_bf16_kernel<<<512, 256, 0, stream>>>(guW3, gblob, 64, 128, NSTEP, W3_OFF, WBLOB_U16);
  gr_pack_bf16_kernel<<<256, 256, 0, stream>>>(jxW2, jblob, 64, 64,  NSTEP, 0,      WBLOB_U16);
  gr_pack_bf16_kernel<<<512, 256, 0, stream>>>(jxW3, jblob, 64, 128, NSTEP, W3_OFF, WBLOB_U16);
  gr_mcjump_kernel<<<RC, 128, 0, stream>>>(u_mc, jump_l, jmn, cr, mcj);

  gr_sim_kernel<<<BATCH / 128, 256, 0, stream>>>(
      rt0, xt0, dBt, u_jump, u_size,
      uW1, ub1, ub2, ub3, uE,
      guW1, gub1, gub2, gub3, guE,
      jxW1, jxb1, jxb2, jxb3,
      jump_r, jump_l, jmn, cr, cfr, mcj,
      uW2p, uW3p, gblob, jblob,
      (float*)d_out);
}
